// PointerGenerator_25460566130981
// MI455X (gfx1250) — matrix operand order repaired and checked by reading, not compiled, not run on hardware
//
#include <hip/hip_runtime.h>

typedef __attribute__((ext_vector_type(16))) _Float16 v16h;
typedef __attribute__((ext_vector_type(8)))  _Float16 v8h;
typedef __attribute__((ext_vector_type(8)))  float    v8f;

#define BB    16
#define S_IN  400
#define S_OUT 100
#define HDIM  256
#define EDIM  128
#define VOC   50000
#define EPSF  1e-10f

// ---------------------------------------------------------------- helpers

__device__ inline float sigf(float x) { return 1.f / (1.f + expf(-x)); }

__device__ inline v8f wmma32(v16h a, v16h b, v8f c) {
  // D = A(16x32 f16) * B(32x16 f16) + C(16x16 f32)
  return __builtin_amdgcn_wmma_f32_16x16x32_f16(false, a, false, b, (short)0, c,
                                                false, false);
}

// A-frag (16x32, K-contiguous rows): lane m=l&15; halves j<8 -> k=kb+hi*8+j,
// j>=8 -> k=kb+16+hi*8+(j-8). Two 16B loads.
__device__ inline v16h ld_a(const _Float16* rp, int kb, int hi) {
  v8h x1 = *(const v8h*)(rp + kb + hi * 8);
  v8h x2 = *(const v8h*)(rp + kb + 16 + hi * 8);
  v16h a;
#pragma unroll
  for (int i = 0; i < 8; ++i) { a[i] = x1[i]; a[i + 8] = x2[i]; }
  return a;
}

// B-frag (32x16, rows are output cols, K-contiguous): lane n=l&15,
// halves j<8 -> k=kb+hi*8+j, j>=8 -> k=kb+16+hi*8+(j-8). Two 16B loads.
__device__ inline v16h ld_b(const _Float16* rp, int kb, int hi) {
  v8h x1 = *(const v8h*)(rp + kb + hi * 8);
  v8h x2 = *(const v8h*)(rp + kb + 16 + hi * 8);
  v16h b;
#pragma unroll
  for (int i = 0; i < 8; ++i) { b[i] = x1[i]; b[i + 8] = x2[i]; }
  return b;
}

__device__ inline v16h ld_a_g(const _Float16* A, int row, int lda, int kb,
                              int K, bool valid, int hi) {
  v16h a;
  if (!valid) {
#pragma unroll
    for (int i = 0; i < 16; ++i) a[i] = (_Float16)0.f;
    return a;
  }
  const _Float16* rp = A + (size_t)row * lda;
  if (kb + 32 <= K) return ld_a(rp, kb, hi);
#pragma unroll
  for (int j = 0; j < 8; ++j) {
    int k1 = kb + hi * 8 + j;
    int k2 = k1 + 16;
    a[j]     = (k1 < K) ? rp[k1] : (_Float16)0.f;
    a[j + 8] = (k2 < K) ? rp[k2] : (_Float16)0.f;
  }
  return a;
}

__device__ inline v16h ld_b_g(const _Float16* B, int row, int ldb, int kb,
                              int K, bool valid, int hi) {
  v16h b;
  if (!valid) {
#pragma unroll
    for (int i = 0; i < 16; ++i) b[i] = (_Float16)0.f;
    return b;
  }
  const _Float16* rp = B + (size_t)row * ldb;
  if (kb + 32 <= K) return ld_b(rp, kb, hi);
#pragma unroll
  for (int j = 0; j < 16; ++j) {
    int k = kb + hi * 8 + (j & 7) + ((j >> 3) << 4);
    b[j] = (k < K) ? rp[k] : (_Float16)0.f;
  }
  return b;
}

// ---------------------------------------------------------------- tiny kernels

__global__ void k_cvt(const float* __restrict__ s, _Float16* __restrict__ d,
                      long n) {
  long i = (long)blockIdx.x * blockDim.x + threadIdx.x;
  long st = (long)gridDim.x * blockDim.x;
  for (; i < n; i += st) d[i] = (_Float16)s[i];
}

__global__ void k_embed(const int* __restrict__ tok,
                        const float* __restrict__ emb,
                        _Float16* __restrict__ o, long rows) {
  long n = rows * EDIM;
  long i = (long)blockIdx.x * blockDim.x + threadIdx.x;
  long st = (long)gridDim.x * blockDim.x;
  for (; i < n; i += st) {
    long r = i >> 7;
    int  e = (int)(i & 127);
    o[i] = (_Float16)emb[(long)tok[r] * EDIM + e];
  }
}

__global__ void k_firstocc(const int* __restrict__ x,
                           unsigned char* __restrict__ f) {
  int b = blockIdx.x, s = threadIdx.x;
  if (s < S_IN) {
    int v = x[b * S_IN + s];
    unsigned char fo = 1;
    for (int s2 = 0; s2 < s; ++s2)
      if (x[b * S_IN + s2] == v) { fo = 0; break; }
    f[b * S_IN + s] = fo;
  }
}

// ---------------------------------------------------------------- persistent LSTM
// One workgroup per LSTM. Wave u owns hidden units [16u,16u+16) for all 16
// batches; cell state c lives in its f32 accumulator fragment across steps.
// h is shared via LDS; per step each wave streams its weight fragments from
// the L2-resident f16 weight buffers through a flattened, one-deep software
// pipeline (input phase chains into recurrent phase). An opaque per-step
// offset blocks LICM from hoisting the weight fragments into registers, which
// at this block size would exceed the VGPR budget and spill to scratch.

template <int G>
__device__ void lstm_body(const _Float16* __restrict__ xemb, int T, bool rev,
                          const _Float16* __restrict__ Wih,
                          const _Float16* __restrict__ Whh,
                          const float* __restrict__ bias,
                          const float* __restrict__ h0,
                          const float* __restrict__ c0,
                          _Float16* __restrict__ outh, int outW, int choff,
                          _Float16* __restrict__ outT,
                          float* __restrict__ hfin, float* __restrict__ cfin) {
  constexpr int NE = EDIM / 32;      // input k-groups (4)
  constexpr int NK = G / 32;         // recurrent k-groups
  constexpr int L  = (NE + NK) * 4;  // weight fragments per step
  __shared__ _Float16 hbuf[16 * G];
  const int tid = threadIdx.x;
  const int lane = tid & 31, u = tid >> 5;
  const int lq = lane & 15, hi = lane >> 4;
  const int unit = 16 * u + lq;

  for (int i = tid; i < 16 * G; i += blockDim.x)
    hbuf[i] = h0 ? (_Float16)h0[i] : (_Float16)0.f;

  float creg[8];
#pragma unroll
  for (int v = 0; v < 8; ++v) {
    int m = v + 8 * hi;
    creg[v] = c0 ? c0[m * G + unit] : 0.f;
  }
  __syncthreads();

  const _Float16* wihRow[4];
  const _Float16* whhRow[4];
  float bval[4];
#pragma unroll
  for (int g = 0; g < 4; ++g) {  // gate order: i, f, g, o
    int row = g * G + unit;
    wihRow[g] = Wih + (size_t)row * EDIM;
    whhRow[g] = Whh + (size_t)row * G;
    bval[g] = bias ? bias[row] : 0.f;
  }
  const _Float16* xrow = xemb + (size_t)lq * T * EDIM;  // batch = lq

  for (int tt = 0; tt < T; ++tt) {
    int t = rev ? (T - 1 - tt) : tt;

    // opaque zero: weight addresses depend on it, so fragment loads cannot be
    // hoisted (and register-spilled) out of the step loop.
    size_t wofs = 0;
    asm volatile("" : "+s"(wofs));

    v8f acc[4];
#pragma unroll
    for (int g = 0; g < 4; ++g)
#pragma unroll
      for (int i = 0; i < 8; ++i) acc[g][i] = bval[g];

    const _Float16* xr = xrow + (size_t)t * EDIM;
    const _Float16* hr = hbuf + lq * G;

    v16h acur = ld_a(xr, 0, hi);                    // group-0 a operand
    v16h anxt = acur;
    v16h wf = ld_b(wihRow[0] + wofs, 0, hi);        // first weight fragment

#pragma unroll
    for (int i = 0; i < L; ++i) {
      const int g = i & 3, grp = i >> 2;
      // one-deep weight prefetch (chains across the phase boundary)
      v16h wn = wf;
      if (i + 1 < L) {
        const int g1 = (i + 1) & 3, grp1 = (i + 1) >> 2;
        wn = (grp1 < NE) ? ld_b(wihRow[g1] + wofs, grp1 * 32, hi)
                         : ld_b(whhRow[g1] + wofs, (grp1 - NE) * 32, hi);
      }
      // prefetch next group's a operand one group (4 WMMAs) ahead
      if (g == 0 && grp + 1 < NE + NK) {
        const int grp1 = grp + 1;
        anxt = (grp1 < NE) ? ld_a(xr, grp1 * 32, hi)
                           : ld_a(hr, (grp1 - NE) * 32, hi);
      }
      acc[g] = wmma32(acur, wf, acc[g]);
      wf = wn;
      if (g == 3) acur = anxt;
    }
    __syncthreads();  // everyone done reading old h

#pragma unroll
    for (int v = 0; v < 8; ++v) {
      float c = sigf(acc[1][v]) * creg[v] + sigf(acc[0][v]) * tanhf(acc[2][v]);
      float h = sigf(acc[3][v]) * tanhf(c);
      creg[v] = c;
      int m = v + 8 * hi;
      hbuf[m * G + unit] = (_Float16)h;
      outh[(size_t)(m * T + t) * outW + choff + unit] = (_Float16)h;
      if (outT)
        outT[((size_t)m * 512 + choff + unit) * (size_t)T + t] = (_Float16)h;
      if (hfin && tt == T - 1) {
        hfin[m * outW + choff + unit] = h;
        cfin[m * outW + choff + unit] = c;
      }
    }
    __syncthreads();
  }
}

__global__ __launch_bounds__(512) void k_enc_lstm(
    const _Float16* xemb, const _Float16* WihF, const _Float16* WhhF,
    const float* bF, const _Float16* WihB, const _Float16* WhhB,
    const float* bB, _Float16* xbi, _Float16* xbiT, float* h0d, float* c0d) {
  int dir = blockIdx.x;  // 0=fwd, 1=bwd run concurrently
  lstm_body<HDIM>(xemb, S_IN, dir != 0, dir ? WihB : WihF, dir ? WhhB : WhhF,
                  dir ? bB : bF, nullptr, nullptr, xbi, 2 * HDIM, dir * HDIM,
                  xbiT, h0d, c0d);
}

__global__ __launch_bounds__(1024) void k_dec_lstm(
    const _Float16* aemb, const _Float16* Wih, const _Float16* Whh,
    const float* b, const float* h0, const float* c0, _Float16* dout) {
  lstm_body<2 * HDIM>(aemb, S_OUT, false, Wih, Whh, b, h0, c0, dout, 2 * HDIM,
                      0, nullptr, nullptr, nullptr);
}

// ---------------------------------------------------------------- generic NT GEMM
// C[M,N] = A[M,K] * B[N,K]^T (+bias[N]); both operands row-major K-contiguous.
// Block 256 thr = 8 waves -> 64x64 tile; wave = 16(M) x 32(N), f32 accum.
// K-loop is register double-buffered: loads for step ks+1 are in flight while
// WMMAs consume step ks.

__global__ __launch_bounds__(256) void k_gemm_nt(
    const _Float16* __restrict__ A, const _Float16* __restrict__ B,
    float* __restrict__ C, const float* __restrict__ bias, int M, int N, int K,
    int lda, int ldb, int ldc, long sA, long sB, long sC) {
  A += (long)blockIdx.z * sA;
  B += (long)blockIdx.z * sB;
  C += (long)blockIdx.z * sC;
  const int tid = threadIdx.x, lane = tid & 31, w = tid >> 5;
  const int lq = lane & 15, hi = lane >> 4;
  const int m0 = blockIdx.y * 64 + (w & 3) * 16;
  const int n0 = blockIdx.x * 64 + (w >> 2) * 32;

  float b0v = 0.f, b1v = 0.f;
  if (bias) {
    if (n0 + lq < N) b0v = bias[n0 + lq];
    if (n0 + 16 + lq < N) b1v = bias[n0 + 16 + lq];
  }
  v8f acc0, acc1;
#pragma unroll
  for (int i = 0; i < 8; ++i) { acc0[i] = b0v; acc1[i] = b1v; }

  const int arow = m0 + lq;
  const bool av = arow < M;
  const int b0row = n0 + lq, b1row = n0 + 16 + lq;
  const bool bv0 = b0row < N, bv1 = b1row < N;
  const int nk = (K + 31) >> 5;

  v16h a  = ld_a_g(A, arow, lda, 0, K, av, hi);
  v16h b0 = ld_b_g(B, b0row, ldb, 0, K, bv0, hi);
  v16h b1 = ld_b_g(B, b1row, ldb, 0, K, bv1, hi);
  for (int ks = 0; ks < nk; ++ks) {
    v16h an = a, b0n = b0, b1n = b1;
    if (ks + 1 < nk) {
      int kb1 = (ks + 1) << 5;
      an  = ld_a_g(A, arow, lda, kb1, K, av, hi);
      b0n = ld_b_g(B, b0row, ldb, kb1, K, bv0, hi);
      b1n = ld_b_g(B, b1row, ldb, kb1, K, bv1, hi);
    }
    acc0 = wmma32(a, b0, acc0);
    acc1 = wmma32(a, b1, acc1);
    a = an; b0 = b0n; b1 = b1n;
  }
#pragma unroll
  for (int v = 0; v < 8; ++v) {
    int m = m0 + v + 8 * hi;
    if (m < M) {
      long rb = (long)m * ldc;
      int n = n0 + lq;
      if (n < N) C[rb + n] = acc0[v];
      n = n0 + 16 + lq;
      if (n < N) C[rb + n] = acc1[v];
    }
  }
}

// ---------------------------------------------------------------- attention softmax

__global__ __launch_bounds__(256) void k_attn_softmax(float* __restrict__ sc,
                                                      _Float16* __restrict__ ah) {
  int r = blockIdx.x;
  float* row = sc + (size_t)r * S_IN;
  __shared__ float red[256];
  int tid = threadIdx.x, bd = blockDim.x;
  float loc = -3.4e38f;
  for (int s = tid; s < S_IN; s += bd) loc = fmaxf(loc, row[s]);
  red[tid] = loc;
  __syncthreads();
  for (int s = bd >> 1; s > 0; s >>= 1) {
    if (tid < s) red[tid] = fmaxf(red[tid], red[tid + s]);
    __syncthreads();
  }
  float mx = red[0];
  __syncthreads();
  loc = 0.f;
  for (int s = tid; s < S_IN; s += bd) loc += expf(row[s] - mx);
  red[tid] = loc;
  __syncthreads();
  for (int s = bd >> 1; s > 0; s >>= 1) {
    if (tid < s) red[tid] += red[tid + s];
    __syncthreads();
  }
  float inv = 1.f / red[0];
  for (int s = tid; s < S_IN; s += bd) {
    float p = expf(row[s] - mx) * inv;
    row[s] = p;
    ah[(size_t)r * S_IN + s] = (_Float16)p;
  }
}

// ---------------------------------------------------------------- feature + pgen

__global__ __launch_bounds__(256) void k_feat_pgen(
    const _Float16* __restrict__ dout, const float* __restrict__ ctx,
    const _Float16* __restrict__ aemb, const float* __restrict__ Wp,
    const float* __restrict__ bp, _Float16* __restrict__ feat,
    float* __restrict__ pgen) {
  int r = blockIdx.x;
  __shared__ float red[256];
  int tid = threadIdx.x, bd = blockDim.x;
  const int D = 2 * HDIM;  // 512
  float loc = 0.f;
  for (int j = tid; j < D; j += bd) {
    _Float16 dv = dout[(size_t)r * D + j];
    float cv = ctx[(size_t)r * D + j];
    feat[(size_t)r * 2 * D + j] = dv;             // feature = [dec_out, context]
    feat[(size_t)r * 2 * D + D + j] = (_Float16)cv;
    loc += Wp[j] * cv + Wp[D + j] * (float)dv;    // pgen_feat = [ctx, dec, emb]
  }
  for (int e = tid; e < EDIM; e += bd)
    loc += Wp[2 * D + e] * (float)aemb[(size_t)r * EDIM + e];
  red[tid] = loc;
  __syncthreads();
  for (int s = bd >> 1; s > 0; s >>= 1) {
    if (tid < s) red[tid] += red[tid + s];
    __syncthreads();
  }
  if (tid == 0) pgen[r] = 1.f / (1.f + expf(-(red[0] + bp[0])));
}

// ---------------------------------------------------------------- final log-mix
// In-place on the logits row: softmax stats, dense log(pgen*p+eps), then fix up
// only the <=400 pointer-touched vocab slots (dedup via first-occurrence).

__global__ __launch_bounds__(512) void k_final(
    float* __restrict__ out, const float* __restrict__ attn,
    const float* __restrict__ pgen, const int* __restrict__ x,
    const unsigned char* __restrict__ focc) {
  int r = blockIdx.x, b = r / S_OUT;
  float* row = out + (size_t)r * VOC;
  __shared__ int sidx[S_IN];
  __shared__ float satt[S_IN], slog[S_IN];
  __shared__ float red[512];
  int tid = threadIdx.x, bd = blockDim.x;
  for (int s = tid; s < S_IN; s += bd) {
    sidx[s] = x[b * S_IN + s];
    satt[s] = attn[(size_t)r * S_IN + s];
  }
  __syncthreads();
  float loc = -3.4e38f;
  for (int v = tid; v < VOC; v += bd) loc = fmaxf(loc, row[v]);
  red[tid] = loc;
  __syncthreads();
  for (int s = bd >> 1; s > 0; s >>= 1) {
    if (tid < s) red[tid] = fmaxf(red[tid], red[tid + s]);
    __syncthreads();
  }
  float mx = red[0];
  __syncthreads();
  loc = 0.f;
  for (int v = tid; v < VOC; v += bd) loc += expf(row[v] - mx);
  red[tid] = loc;
  __syncthreads();
  for (int s = bd >> 1; s > 0; s >>= 1) {
    if (tid < s) red[tid] += red[tid + s];
    __syncthreads();
  }
  float inv = 1.f / red[0];
  __syncthreads();
  for (int s = tid; s < S_IN; s += bd) slog[s] = row[sidx[s]];  // save logits
  __syncthreads();
  float pg = pgen[r];
  for (int v = tid; v < VOC; v += bd)
    row[v] = logf(pg * expf(row[v] - mx) * inv + EPSF);
  __threadfence();
  __syncthreads();
  for (int s = tid; s < S_IN; s += bd) {
    if (focc[b * S_IN + s]) {
      int id = sidx[s];
      float tot = 0.f;
      for (int s2 = 0; s2 < S_IN; ++s2)
        if (sidx[s2] == id) tot += satt[s2];
      row[id] = logf(pg * expf(slog[s] - mx) * inv + (1.f - pg) * tot + EPSF);
    }
  }
}

// ---------------------------------------------------------------- launch

extern "C" void kernel_launch(void* const* d_in, const int* in_sizes, int n_in,
                              void* d_out, int out_size, void* d_ws,
                              size_t ws_size, hipStream_t stream) {
  (void)in_sizes; (void)n_in; (void)out_size;
  const int*   x    = (const int*)d_in[0];
  const int*   ans  = (const int*)d_in[1];
  const float* emb  = (const float*)d_in[2];
  const float* WihF = (const float*)d_in[3];
  const float* WhhF = (const float*)d_in[4];
  const float* bF   = (const float*)d_in[5];
  const float* WihB = (const float*)d_in[6];
  const float* WhhB = (const float*)d_in[7];
  const float* bB   = (const float*)d_in[8];
  const float* dWih = (const float*)d_in[9];
  const float* dWhh = (const float*)d_in[10];
  const float* db   = (const float*)d_in[11];
  const float* Wpro = (const float*)d_in[12];
  const float* bpro = (const float*)d_in[13];
  const float* Wpg  = (const float*)d_in[14];
  const float* bpg  = (const float*)d_in[15];
  float* out = (float*)d_out;

  char* wsb = (char*)d_ws;
  size_t off = 0;
  auto carve = [&](size_t bytes) -> void* {
    void* p = wsb + off;
    off = (off + bytes + 255) & ~(size_t)255;
    return p;
  };
  const size_t nWih  = 4 * HDIM * EDIM;            // 131072
  const size_t nWhh  = 4 * HDIM * HDIM;            // 262144
  const size_t nDWih = 8 * HDIM * EDIM;            // 262144
  const size_t nDWhh = (size_t)8 * HDIM * 2 * HDIM;// 1048576
  const size_t nWpro = (size_t)VOC * 4 * HDIM;     // 51.2M
  _Float16* wihF_h = (_Float16*)carve(nWih * 2);
  _Float16* whhF_h = (_Float16*)carve(nWhh * 2);
  _Float16* wihB_h = (_Float16*)carve(nWih * 2);
  _Float16* whhB_h = (_Float16*)carve(nWhh * 2);
  _Float16* dwih_h = (_Float16*)carve(nDWih * 2);
  _Float16* dwhh_h = (_Float16*)carve(nDWhh * 2);
  _Float16* wpro_h = (_Float16*)carve(nWpro * 2);
  _Float16* xemb_h = (_Float16*)carve((size_t)BB * S_IN * EDIM * 2);
  _Float16* aemb_h = (_Float16*)carve((size_t)BB * S_OUT * EDIM * 2);
  _Float16* xbi_h  = (_Float16*)carve((size_t)BB * S_IN * 512 * 2);
  _Float16* xbiT_h = (_Float16*)carve((size_t)BB * 512 * S_IN * 2);
  float*    h0d    = (float*)carve((size_t)BB * 512 * 4);
  float*    c0d    = (float*)carve((size_t)BB * 512 * 4);
  _Float16* dout_h = (_Float16*)carve((size_t)BB * S_OUT * 512 * 2);
  float*    scores = (float*)carve((size_t)BB * S_OUT * S_IN * 4);
  _Float16* attn_h = (_Float16*)carve((size_t)BB * S_OUT * S_IN * 2);
  float*    ctx    = (float*)carve((size_t)BB * S_OUT * 512 * 4);
  _Float16* feat_h = (_Float16*)carve((size_t)BB * S_OUT * 1024 * 2);
  float*    pgenf  = (float*)carve((size_t)BB * S_OUT * 4);
  unsigned char* focc = (unsigned char*)carve((size_t)BB * S_IN);
  if (off > ws_size) return;  // workspace too small: bail safely

  auto cvt = [&](const float* s, _Float16* d, size_t n) {
    int blocks = (int)((n + 255) / 256);
    if (blocks > 4096) blocks = 4096;
    k_cvt<<<blocks, 256, 0, stream>>>(s, d, (long)n);
  };
  cvt(WihF, wihF_h, nWih);
  cvt(WhhF, whhF_h, nWhh);
  cvt(WihB, wihB_h, nWih);
  cvt(WhhB, whhB_h, nWhh);
  cvt(dWih, dwih_h, nDWih);
  cvt(dWhh, dwhh_h, nDWhh);
  cvt(Wpro, wpro_h, nWpro);

  k_embed<<<1024, 256, 0, stream>>>(x, emb, xemb_h, (long)BB * S_IN);
  k_embed<<<512, 256, 0, stream>>>(ans, emb, aemb_h, (long)BB * S_OUT);

  // encoder: fwd + bwd as two concurrent persistent workgroups
  k_enc_lstm<<<2, 512, 0, stream>>>(xemb_h, wihF_h, whhF_h, bF, wihB_h, whhB_h,
                                    bB, xbi_h, xbiT_h, h0d, c0d);
  // decoder: one persistent workgroup, hidden 512
  k_dec_lstm<<<1, 1024, 0, stream>>>(aemb_h, dwih_h, dwhh_h, db, h0d, c0d,
                                     dout_h);

  // scores[b,t,s] = dec_out . x_bilstm  (batched NT, K=512)
  k_gemm_nt<<<dim3((S_IN + 63) / 64, (S_OUT + 63) / 64, BB), 256, 0, stream>>>(
      dout_h, xbi_h, scores, nullptr, S_OUT, S_IN, 512, 512, 512, S_IN,
      (long)S_OUT * 512, (long)S_IN * 512, (long)S_OUT * S_IN);

  k_attn_softmax<<<BB * S_OUT, 256, 0, stream>>>(scores, attn_h);

  // context = attn @ x_bilstm  (batched NT against transposed bilstm, K=400)
  k_gemm_nt<<<dim3((512 + 63) / 64, (S_OUT + 63) / 64, BB), 256, 0, stream>>>(
      attn_h, xbiT_h, ctx, nullptr, S_OUT, 512, S_IN, S_IN, S_IN, 512,
      (long)S_OUT * S_IN, (long)512 * S_IN, (long)S_OUT * 512);

  k_feat_pgen<<<BB * S_OUT, 256, 0, stream>>>(dout_h, ctx, aemb_h, Wpg, bpg,
                                              feat_h, pgenf);

  // vocab projection: [1600,1024] x [50000,1024]^T -> d_out (+b_pro)
  k_gemm_nt<<<dim3((VOC + 63) / 64, (BB * S_OUT + 63) / 64, 1), 256, 0,
              stream>>>(feat_h, wpro_h, out, bpro, BB * S_OUT, VOC, 1024, 1024,
                        1024, VOC, 0, 0, 0);

  k_firstocc<<<BB, 512, 0, stream>>>(x, focc);
  k_final<<<BB * S_OUT, 512, 0, stream>>>(out, scores, pgenf, x, focc);
}
